// GCNBlock_55138790146460
// MI455X (gfx1250) — matrix operand order repaired and checked by reading, not compiled, not run on hardware
//
#include <hip/hip_runtime.h>
#include <hip/hip_bf16.h>

// ---------------------------------------------------------------------------
// CTR-GCN block for MI455X (gfx1250, wave32, WMMA).
//
// Roofline: ~14 GFLOP, ~220MB mandatory HBM traffic (x in + outputs out)
// -> memory/balanced at 23.3 TB/s. Strategy: every 1x1 conv and the
// adjacency einsum run as v_wmma_f32_16x16x32_f16 GEMMs with f32 accum;
// intermediates are f16 in workspace (~138MB, mostly resident in 192MB L2);
// BN/bias/ReLU/residual fused into GEMM epilogues.
//
// WMMA register layouts per CDNA5 ISA 7.12.2:
//   A (16x32 f16): lane l<16 -> row l, K {0..7,16..23}; lane l>=16 -> K {8..15,24..31}
//   B (32x16 f16): lane l -> col l&15, K same as A {0..7,16..23} / {8..15,24..31}  (two 8-half runs
//       when B is stored transposed [out][k] == conv weight layout (O,I))
//   C/D (16x16 f32): VGPR r -> row r + 8*(lane>>4), col lane&15
//
// Round-1 fixes vs round-0:
//  * front GEMM output groups padded to 32 channels (O=96): destination buffer
//    and ReLU flag are wave-uniform per tile -> no 3-way exec-mask divergence.
//  * all front outputs stored at linear spatial index p -> zero div/mod-by-25
//    in epilogues; v->32 padding moved into K4's (cheap) LDS staging.
//  * LDS row strides padded (72 / 40 halves) to break bank conflicts while
//    keeping 16B alignment for ds_load_b128 fragments.
// ---------------------------------------------------------------------------

typedef _Float16 v16h __attribute__((ext_vector_type(16)));
typedef _Float16 v8h  __attribute__((ext_vector_type(8)));
typedef float    v8f  __attribute__((ext_vector_type(8)));
typedef _Float16 half_t;
typedef unsigned int u32;

#define N_   128
#define C_   64
#define T_   128
#define V_   25
#define K_   3
#define MID_ 8
#define KM_  24            // K_*MID_
#define VG_  26            // V_+1 (with appended v-mean column)
#define PTV  3200          // T_*V_
#define PTVG 3328          // T_*VG_
#define LDA64 72           // LDS row stride (halves) for K=64 tiles
#define LDA32 40           // LDS row stride (halves) for K=32 tiles

// ---- flattened input-pointer table ----------------------------------------
// setup_inputs order: x first, then params pytree leaves (jax tree order:
// dict keys sorted alphabetically at each level; conv={b,w}; bn={b,g,m,v}).
enum {
  IN_X = 0,
  G_A = 1, G_ALPHA, G_BETA, G_C1_B, G_C1_W, G_C2_B, G_C2_W,
  G_OBN_B, G_OBN_G, G_OBN_M, G_OBN_V,
  G_POST_B, G_POST_W, G_PRE_B, G_PRE_W,
  G_PBN_B, G_PBN_G, G_PBN_M, G_PBN_V,           // ..19
  T_ADD = 20,
  T_BR0 = 21,   // per branch b (12 leaves): +0..3 bn1{b,g,m,v}, +4..7 bn2{b,g,m,v}, +8 c1.b, +9 c1.w, +10 ct.b, +11 ct.w
  T_C1X1_B = 69, T_C1X1_W = 70,
  T_MB_BN_B = 71, T_MB_BN_G = 72, T_MB_BN_M = 73, T_MB_BN_V = 74,
  T_MB_C1_B = 75, T_MB_C1_W = 76,
  T_OBN_B = 77, T_OBN_G = 78, T_OBN_M = 79, T_OBN_V = 80,
  T_TBN_B = 81, T_TBN_G = 82, T_TBN_M = 83, T_TBN_V = 84,
  T_TCONV_B = 85, T_TCONV_W = 86,
  NUM_IN = 87
};

struct PtrTab { const float* p[NUM_IN]; };

struct Pack {
  half_t *Wh1, *Wh5, *Wh7, *Wh9;        // f16 weights, transposed [O][Kpad]
  float *s1, *h1;                       // front GEMM per-channel affine (96)
  float *s5, *h5;                       // post GEMM affine (post.b + gcn out_bn)
  float *s7, *h7;                       // branch GEMM affine (c1.b + bn1 / bias)
  float *ctw, *sct, *hct;               // temporal conv weights [44][14][3] + affine
  float *st, *ht;                       // t_bn affine
  float *s9, *h9;                       // t_conv.b + tcn out_bn affine
};

// ---------------------------------------------------------------------------
// WMMA helpers
// ---------------------------------------------------------------------------
__device__ __forceinline__ v8f wmma32(v16h a, v16h b, v8f c) {
  // (neg_a, A, neg_b, B, c_mod, C, reuse_a, reuse_b)
  return __builtin_amdgcn_wmma_f32_16x16x32_f16(false, a, false, b, (short)0, c, false, false);
}
// A fragment from LDS row (row-major, row points at row start + ks*32)
__device__ __forceinline__ v16h lda(const half_t* row, int lhi) {
  const int kb = lhi << 3;
  v8h lo = *(const v8h*)(row + kb);
  v8h hi = *(const v8h*)(row + kb + 16);
  return __builtin_shufflevector(lo, hi, 0,1,2,3,4,5,6,7,8,9,10,11,12,13,14,15);
}
// B fragment straight from global (Bt stored [O][Kpad], col points at o*Kpad + ks*32)
__device__ __forceinline__ v16h ldb(const half_t* col, int lhi) {
  const int kb = lhi << 3;
  v8h lo = *(const v8h*)(col + kb);
  v8h hi = *(const v8h*)(col + kb + 16);
  return __builtin_shufflevector(lo, hi, 0,1,2,3,4,5,6,7,8,9,10,11,12,13,14,15);
}

__device__ __forceinline__ void br_of(int gco, int& off, int& ch, int& dil) {
  if (gco < 14)      { off = 0;  ch = 14; dil = 1; }
  else if (gco < 24) { off = 14; ch = 10; dil = 2; }
  else if (gco < 34) { off = 24; ch = 10; dil = 3; }
  else               { off = 34; ch = 10; dil = 4; }
}

// ---------------------------------------------------------------------------
// K0: pack weights (f16, transposed) and fold conv-bias + BN into per-channel
//     affine (scale, shift). Single block; tiny.
// ---------------------------------------------------------------------------
__global__ void k_pack(PtrTab P, Pack o) {
  const int tid = threadIdx.x, NTH = blockDim.x;
  const float EPS = 1e-5f;

  // front GEMM: 3 groups of 32 rows {pre(24)+pad, c1(24)+pad, c2(24)+pad}, K=64
  for (int i = tid; i < 96 * 64; i += NTH) {
    int r = i >> 6, k = i & 63, g = r >> 5, lo = r & 31;
    float w = 0.f;
    if (lo < 24) {
      if (g == 0)      w = P.p[G_PRE_W][lo * 64 + k];
      else if (g == 1) w = P.p[G_C1_W][lo * 64 + k];
      else             w = P.p[G_C2_W][lo * 64 + k];
    }
    o.Wh1[i] = (half_t)w;
  }
  for (int i = tid; i < 96; i += NTH) {
    int g = i >> 5, lo = i & 31;
    float s = 0.f, h = 0.f;
    if (lo < 24) {
      if (g == 0) {
        float sc = P.p[G_PBN_G][lo] * rsqrtf(P.p[G_PBN_V][lo] + EPS);
        s = sc; h = P.p[G_PRE_B][lo] * sc + P.p[G_PBN_B][lo] - P.p[G_PBN_M][lo] * sc;
      } else if (g == 1) { s = 1.f; h = P.p[G_C1_B][lo]; }
      else               { s = 1.f; h = P.p[G_C2_B][lo]; }
    }
    o.s1[i] = s; o.h1[i] = h;
  }
  // post GEMM: [64][32] (K=24 padded to 32)
  for (int i = tid; i < 64 * 32; i += NTH) {
    int r = i >> 5, k = i & 31;
    o.Wh5[i] = (half_t)((k < 24) ? P.p[G_POST_W][r * 24 + k] : 0.f);
  }
  for (int i = tid; i < 64; i += NTH) {
    float sc = P.p[G_OBN_G][i] * rsqrtf(P.p[G_OBN_V][i] + EPS);
    o.s5[i] = sc;
    o.h5[i] = P.p[G_POST_B][i] * sc + P.p[G_OBN_B][i] - P.p[G_OBN_M][i] * sc;
  }
  // branch GEMM: rows = {b0(14),b1(10),b2(10),b3(10),maxb(10),c1x1(10)}, K=64
  for (int i = tid; i < 64 * 64; i += NTH) {
    int r = i >> 6, k = i & 63; const float* w; int lr;
    if (r < 14)      { w = P.p[T_BR0 + 0 * 12 + 9]; lr = r; }
    else if (r < 24) { w = P.p[T_BR0 + 1 * 12 + 9]; lr = r - 14; }
    else if (r < 34) { w = P.p[T_BR0 + 2 * 12 + 9]; lr = r - 24; }
    else if (r < 44) { w = P.p[T_BR0 + 3 * 12 + 9]; lr = r - 34; }
    else if (r < 54) { w = P.p[T_MB_C1_W];          lr = r - 44; }
    else             { w = P.p[T_C1X1_W];           lr = r - 54; }
    o.Wh7[i] = (half_t)w[lr * 64 + k];
  }
  for (int i = tid; i < 64; i += NTH) {
    float s, h;
    if (i < 44) {
      int off, ch, dil; br_of(i, off, ch, dil);
      int b = (i < 14) ? 0 : (i < 24) ? 1 : (i < 34) ? 2 : 3;
      int base = T_BR0 + b * 12; int lr = i - off;
      float sc = P.p[base + 1][lr] * rsqrtf(P.p[base + 3][lr] + EPS);
      s = sc; h = P.p[base + 8][lr] * sc + P.p[base + 0][lr] - P.p[base + 2][lr] * sc;
    } else if (i < 54) {
      int lr = i - 44;
      float sc = P.p[T_MB_BN_G][lr] * rsqrtf(P.p[T_MB_BN_V][lr] + EPS);
      s = sc; h = P.p[T_MB_C1_B][lr] * sc + P.p[T_MB_BN_B][lr] - P.p[T_MB_BN_M][lr] * sc;
    } else { s = 1.f; h = P.p[T_C1X1_B][i - 54]; }
    o.s7[i] = s; o.h7[i] = h;
  }
  // temporal conv weights [44][14][3] (ci padded to 14 with zeros)
  for (int i = tid; i < 44 * 14 * 3; i += NTH) {
    int gco = i / 42, ci = (i / 3) % 14, kt = i % 3;
    int off, ch, dil; br_of(gco, off, ch, dil);
    int b = (gco < 14) ? 0 : (gco < 24) ? 1 : (gco < 34) ? 2 : 3;
    int co = gco - off; float w = 0.f;
    if (ci < ch) w = P.p[T_BR0 + b * 12 + 11][(co * ch + ci) * 3 + kt];
    o.ctw[i] = w;
  }
  for (int i = tid; i < 44; i += NTH) {
    int off, ch, dil; br_of(i, off, ch, dil);
    int b = (i < 14) ? 0 : (i < 24) ? 1 : (i < 34) ? 2 : 3;
    int base = T_BR0 + b * 12; int co = i - off;
    float sc = P.p[base + 5][co] * rsqrtf(P.p[base + 7][co] + EPS);
    o.sct[i] = sc;
    o.hct[i] = P.p[base + 10][co] * sc + P.p[base + 4][co] - P.p[base + 6][co] * sc;
  }
  // t_bn (no conv bias folded)
  for (int i = tid; i < 64; i += NTH) {
    float sc = P.p[T_TBN_G][i] * rsqrtf(P.p[T_TBN_V][i] + EPS);
    o.st[i] = sc; o.ht[i] = P.p[T_TBN_B][i] - P.p[T_TBN_M][i] * sc;
  }
  // t_conv GEMM + tcn out_bn
  for (int i = tid; i < 64 * 64; i += NTH) o.Wh9[i] = (half_t)P.p[T_TCONV_W][i];
  for (int i = tid; i < 64; i += NTH) {
    float sc = P.p[T_OBN_G][i] * rsqrtf(P.p[T_OBN_V][i] + EPS);
    o.s9[i] = sc;
    o.h9[i] = P.p[T_TCONV_B][i] * sc + P.p[T_OBN_B][i] - P.p[T_OBN_M][i] * sc;
  }
}

// ---------------------------------------------------------------------------
// K1: fused pre/c1/c2 GEMM.  M=N*T*V, K=64, O=3x32 (24 live per group).
//     All outputs stored compact at linear spatial index p (f16):
//       fo[g][(n*24+ol)*PTV + p],  g in {pre, c1, c2}.
//     Group (=dest buffer + relu flag) is wave-uniform per tile.
// ---------------------------------------------------------------------------
__global__ __launch_bounds__(256) void k_gemm_front(
    const float* __restrict__ x, const half_t* __restrict__ Wh1,
    const float* __restrict__ s1, const float* __restrict__ h1,
    half_t* __restrict__ fo) {
  __shared__ __align__(64) half_t As[128 * LDA64];
  const int tid = threadIdx.x;
  const int n = blockIdx.x / 25;
  const int p0 = (blockIdx.x % 25) * 128;       // 3200 rows per n, 25 blocks per n
  const float* xb = x + n * 64 * PTV + p0;
  for (int i = tid; i < 128 * 64; i += 256) {   // coalesced along m (t*25+v contiguous)
    int ml = i & 127, c = i >> 7;
    As[ml * LDA64 + c] = (half_t)xb[c * PTV + ml];
  }
  __builtin_prefetch(Wh1, 0, 2);                // global_prefetch_b8 on gfx1250
  __syncthreads();
  const int lane = tid & 31, wid = tid >> 5, lhi = lane >> 4, llo = lane & 15;
  const size_t SL = (size_t)N_ * KM_ * PTV;     // group slice
  for (int ti = wid; ti < 48; ti += 8) {        // 8 m-tiles x 6 n-tiles, 8 waves
    const int mt = ti & 7, nt = ti >> 3;
    v8f acc{};
#pragma unroll
    for (int ks = 0; ks < 2; ++ks) {
      v16h a = lda(&As[(mt * 16 + llo) * LDA64 + ks * 32], lhi);
      v16h b = ldb(Wh1 + (nt * 16 + llo) * 64 + ks * 32, lhi);
      acc = wmma32(a, b, acc);
    }
    const int o = nt * 16 + llo;                // 0..95 (row in packed W)
    const int g = nt >> 1;                      // group 0..2 (wave-uniform)
    const int ol = ((nt & 1) << 4) + llo;       // channel within group, 0..31
    const bool live = (ol < 24);
    const float s = s1[o], h = h1[o];
    half_t* dst = fo + g * SL + (size_t)(n * KM_ + (ol < 24 ? ol : 0)) * PTV;
#pragma unroll
    for (int r = 0; r < 8; ++r) {
      int p = p0 + mt * 16 + r + (lhi << 3);
      float val = acc[r] * s + h;
      if (g == 0) val = fmaxf(val, 0.f);        // pre: bn+relu (uniform branch)
      if (live) dst[p] = (half_t)val;
    }
  }
}

// K2: means over t of c1/c2 -> x1m, x2m  (n,24,25) each
__global__ void k_mean_t(const half_t* __restrict__ c1raw, const half_t* __restrict__ c2raw,
                         float* __restrict__ x1m, float* __restrict__ x2m) {
  int i = blockIdx.x * blockDim.x + threadIdx.x;
  if (i >= 2 * N_ * KM_ * V_) return;
  int sel = i / (N_ * KM_ * V_), j = i % (N_ * KM_ * V_);
  int nc = j / V_, v = j % V_;
  const half_t* src = sel ? c2raw : c1raw;
  float s = 0.f;
  for (int t = 0; t < T_; ++t) s += (float)src[nc * PTV + t * V_ + v];
  (sel ? x2m : x1m)[j] = s * (1.f / T_);
}

// ---------------------------------------------------------------------------
// K3: attention. Per (n,k): scores = x1m . x2m over MID, softmax over v,
//     graph = tanh(x1-x2)*alpha0 + intra (output 2), afull = graph + A
//     stored transposed+padded [nk][w(32)][v(32)] f16 as the B operand of K4.
// ---------------------------------------------------------------------------
__global__ __launch_bounds__(256) void k_attn(
    const float* __restrict__ x1m, const float* __restrict__ x2m,
    const float* __restrict__ A, const float* __restrict__ alpha,
    const float* __restrict__ beta,
    half_t* __restrict__ afull, float* __restrict__ graph) {
  __shared__ float x1s[MID_ * V_], x2s[MID_ * V_], sc[V_ * V_];
  const int tid = threadIdx.x;
  const int n = blockIdx.x / K_, k = blockIdx.x % K_;
  const float a0 = alpha[0], b0 = beta[0];
  for (int i = tid; i < MID_ * V_; i += 256) {
    int cc = i / V_, v = i % V_;
    int g = (n * KM_ + k * MID_ + cc) * V_ + v;
    x1s[i] = x1m[g]; x2s[i] = x2m[g];
  }
  __syncthreads();
  for (int i = tid; i < V_ * V_; i += 256) {
    int v = i / V_, w = i % V_; float s = 0.f;
#pragma unroll
    for (int cc = 0; cc < MID_; ++cc) s += x1s[cc * V_ + v] * x2s[cc * V_ + w];
    sc[i] = s;
  }
  __syncthreads();
  if (tid < V_) {                               // softmax over v (axis=-2), per column w
    int w = tid;
    float m = -1e30f;
    for (int v = 0; v < V_; ++v) m = fmaxf(m, sc[v * V_ + w]);
    float sum = 0.f;
    for (int v = 0; v < V_; ++v) sum += expf(sc[v * V_ + w] - m);
    float inv = b0 / sum;
    for (int v = 0; v < V_; ++v) sc[v * V_ + w] = expf(sc[v * V_ + w] - m) * inv;
  }
  __syncthreads();
  for (int i = tid; i < MID_ * 32 * 32; i += 256) {
    int cc = i >> 10, rem = i & 1023, w = rem >> 5, v = rem & 31;
    int nk = n * KM_ + k * MID_ + cc;
    if (v < V_ && w < V_) {
      float g = tanhf(x1s[cc * V_ + v] - x2s[cc * V_ + w]) * a0 + sc[v * V_ + w];
      graph[(nk * V_ + v) * V_ + w] = g;
      afull[(nk << 10) + w * 32 + v] = (half_t)(g + A[(k * V_ + v) * V_ + w]);
    } else {
      afull[(nk << 10) + w * 32 + v] = (half_t)0.f;   // K-dim padding must be 0
    }
  }
}

// ---------------------------------------------------------------------------
// K4: adjacency einsum  Y[t,w] = pre[t,v] . afull[v,w]  per (n,kc).
//     K = v = 32 (v>=25 zero-padded during LDS staging; dest-indexed so no
//     integer division anywhere). One WMMA per 16x16 tile.
//     Output stored row-major [m][24] f16 as A operand of K5.
// ---------------------------------------------------------------------------
__global__ __launch_bounds__(128) void k_gemm_adj(
    const half_t* __restrict__ pre_f, const half_t* __restrict__ afull,
    half_t* __restrict__ ycat) {
  __shared__ __align__(64) half_t As[128 * LDA32];
  const int tid = threadIdx.x;
  const int nk = blockIdx.x;                    // n*24 + kc
  const int n = nk / KM_, kc = nk % KM_;
  const half_t* src = pre_f + (size_t)nk * PTV;
  for (int i = tid; i < 128 * 32; i += 128) {
    int t = i >> 5, v = i & 31;
    As[t * LDA32 + v] = (v < V_) ? src[t * V_ + v] : (half_t)0.f;
  }
  __syncthreads();
  const int lane = tid & 31, wid = tid >> 5, lhi = lane >> 4, llo = lane & 15;
  const half_t* bt = afull + (nk << 10);
  for (int ti = wid; ti < 16; ti += 4) {        // 8 m-tiles x 2 n-tiles, 4 waves
    int mt = ti & 7, nt = ti >> 3;
    v16h a = lda(&As[(mt * 16 + llo) * LDA32], lhi);
    v16h b = ldb(bt + (nt * 16 + llo) * 32, lhi);
    v8f acc{};
    acc = wmma32(a, b, acc);
    int w = nt * 16 + llo;
    if (w < V_) {
#pragma unroll
      for (int r = 0; r < 8; ++r) {
        int t = mt * 16 + r + (lhi << 3);
        ycat[(size_t)(n * PTV + t * V_ + w) * KM_ + kc] = (half_t)acc[r];
      }
    }
  }
}

// ---------------------------------------------------------------------------
// K5: post GEMM (24->64) + bias + gcn out_bn + residual(x) + relu -> g (f16)
// ---------------------------------------------------------------------------
__global__ __launch_bounds__(256) void k_gemm_post(
    const half_t* __restrict__ ycat, const half_t* __restrict__ Wh5,
    const float* __restrict__ s5, const float* __restrict__ h5,
    const float* __restrict__ x, half_t* __restrict__ g_h) {
  __shared__ __align__(64) half_t As[128 * LDA32];
  const int tid = threadIdx.x;
  const int mBase = blockIdx.x * 128;
  const int n = blockIdx.x / 25;
  const int p0 = (blockIdx.x % 25) * 128;
  for (int i = tid; i < 128 * 32; i += 256) {
    int ml = i >> 5, kk = i & 31;
    As[ml * LDA32 + kk] = (kk < 24) ? ycat[(size_t)(mBase + ml) * KM_ + kk] : (half_t)0.f;
  }
  __syncthreads();
  const int lane = tid & 31, wid = tid >> 5, lhi = lane >> 4, llo = lane & 15;
  for (int ti = wid; ti < 32; ti += 8) {
    int mt = ti & 7, nt = ti >> 3;
    v16h a = lda(&As[(mt * 16 + llo) * LDA32], lhi);
    v16h b = ldb(Wh5 + (nt * 16 + llo) * 32, lhi);
    v8f acc{};
    acc = wmma32(a, b, acc);
    int o = nt * 16 + llo;
    float s = s5[o], h = h5[o];
#pragma unroll
    for (int r = 0; r < 8; ++r) {
      int p = p0 + mt * 16 + r + (lhi << 3);
      float val = acc[r] * s + h + x[(n * 64 + o) * PTV + p];
      g_h[(n * 64 + o) * PTV + p] = (half_t)fmaxf(val, 0.f);
    }
  }
}

// K6: v-mean of g -> gmean (n,64,t)
__global__ void k_mean_v(const half_t* __restrict__ g_h, float* __restrict__ gmean) {
  int i = blockIdx.x * blockDim.x + threadIdx.x;
  if (i >= N_ * C_ * T_) return;
  int nc = i / T_, t = i % T_;
  float s = 0.f;
  for (int v = 0; v < V_; ++v) s += (float)g_h[nc * PTV + t * V_ + v];
  gmean[i] = s * (1.f / V_);
}

// ---------------------------------------------------------------------------
// K7: all six branch 1x1 convs fused: M=N*T*26 (col 25 = v-mean), K=64, O=64.
//     Per-channel epilogue: branches/maxb -> bn1+relu; c1x1 -> bias only.
// ---------------------------------------------------------------------------
__global__ __launch_bounds__(256) void k_gemm_branch(
    const half_t* __restrict__ g_h, const float* __restrict__ gmean,
    const half_t* __restrict__ Wh7, const float* __restrict__ s7,
    const float* __restrict__ h7, half_t* __restrict__ hbuf) {
  __shared__ __align__(64) half_t As[128 * LDA64];
  const int tid = threadIdx.x;
  const int n = blockIdx.x / 26;                // 3328 rows per n, 26 blocks per n
  const int p0 = (blockIdx.x % 26) * 128;
  for (int i = tid; i < 128 * 64; i += 256) {
    int ml = i & 127, c = i >> 7;
    int p = p0 + ml, t = p / VG_, j = p % VG_;
    float v = (j < V_) ? (float)g_h[(n * 64 + c) * PTV + t * V_ + j]
                       : gmean[(n * 64 + c) * T_ + t];
    As[ml * LDA64 + c] = (half_t)v;
  }
  __syncthreads();
  const int lane = tid & 31, wid = tid >> 5, lhi = lane >> 4, llo = lane & 15;
  for (int ti = wid; ti < 32; ti += 8) {
    int mt = ti & 7, nt = ti >> 3;
    v8f acc{};
#pragma unroll
    for (int ks = 0; ks < 2; ++ks) {
      v16h a = lda(&As[(mt * 16 + llo) * LDA64 + ks * 32], lhi);
      v16h b = ldb(Wh7 + (nt * 16 + llo) * 64 + ks * 32, lhi);
      acc = wmma32(a, b, acc);
    }
    int o = nt * 16 + llo;
    float s = s7[o], h = h7[o];
#pragma unroll
    for (int r = 0; r < 8; ++r) {
      int p = p0 + mt * 16 + r + (lhi << 3);
      float val = acc[r] * s + h;
      if (o < 54) val = fmaxf(val, 0.f);
      hbuf[(n * 64 + o) * PTVG + p] = (half_t)val;
    }
  }
}

// ---------------------------------------------------------------------------
// K8: temporal stage. Per (n,c,t): dilated 3-tap conv + bn2 (c<44),
//     time max-pool (44<=c<54), passthrough (c>=54); then mix in the global
//     column (j=25) with add_coeff, t_bn affine, relu -> feat (f16).
// ---------------------------------------------------------------------------
__global__ __launch_bounds__(256) void k_temporal(
    const half_t* __restrict__ hbuf, const float* __restrict__ ctw,
    const float* __restrict__ sct, const float* __restrict__ hct,
    const float* __restrict__ st, const float* __restrict__ ht,
    const float* __restrict__ addc, half_t* __restrict__ feat) {
  int i = blockIdx.x * blockDim.x + threadIdx.x;
  if (i >= N_ * C_ * T_) return;
  int t = i % T_, c = (i / T_) % C_, n = i / (C_ * T_);
  const half_t* hb = hbuf + (size_t)(n * 64) * PTVG;
  int cls, off = 0, dil = 1, chh = 14;
  float sc_ = 0.f, hc_ = 0.f;
  if (c < 44) { cls = 0; br_of(c, off, chh, dil); sc_ = sct[c]; hc_ = hct[c]; }
  else if (c < 54) cls = 1;
  else cls = 2;

  auto colval = [&](int j) -> float {
    if (cls == 0) {
      float acc = 0.f;
      for (int kt = 0; kt < 3; ++kt) {
        int tt = t + (kt - 1) * dil;
        if ((unsigned)tt < (unsigned)T_) {
          for (int ci = 0; ci < 14; ++ci)       // zero-padded weights past chh
            acc += ctw[(c * 14 + ci) * 3 + kt] *
                   (float)hb[(off + ci) * PTVG + tt * VG_ + j];
        }
      }
      return acc * sc_ + hc_;
    } else if (cls == 1) {
      float m = -1e30f;
      for (int dt = -1; dt <= 1; ++dt) {
        int tt = t + dt;
        if ((unsigned)tt < (unsigned)T_)
          m = fmaxf(m, (float)hb[c * PTVG + tt * VG_ + j]);
      }
      return m;
    }
    return (float)hb[c * PTVG + t * VG_ + j];
  };

  float gv = colval(V_);                        // global column j=25
  float ss = st[c], hh = ht[c];
  for (int j = 0; j < V_; ++j) {
    float lv = colval(j) + gv * addc[j];
    feat[(size_t)(n * 64 + c) * PTV + t * V_ + j] = (half_t)fmaxf(lv * ss + hh, 0.f);
  }
}

// ---------------------------------------------------------------------------
// K9: t_conv GEMM (64->64) + bias + tcn out_bn + residual(x) + relu -> out f32
// ---------------------------------------------------------------------------
__global__ __launch_bounds__(256) void k_gemm_tconv(
    const half_t* __restrict__ feat, const half_t* __restrict__ Wh9,
    const float* __restrict__ s9, const float* __restrict__ h9,
    const float* __restrict__ x, float* __restrict__ out) {
  __shared__ __align__(64) half_t As[128 * LDA64];
  const int tid = threadIdx.x;
  const int n = blockIdx.x / 25;
  const int p0 = (blockIdx.x % 25) * 128;
  for (int i = tid; i < 128 * 64; i += 256) {
    int ml = i & 127, c = i >> 7;
    As[ml * LDA64 + c] = feat[(n * 64 + c) * PTV + p0 + ml];
  }
  __syncthreads();
  const int lane = tid & 31, wid = tid >> 5, lhi = lane >> 4, llo = lane & 15;
  for (int ti = wid; ti < 32; ti += 8) {
    int mt = ti & 7, nt = ti >> 3;
    v8f acc{};
#pragma unroll
    for (int ks = 0; ks < 2; ++ks) {
      v16h a = lda(&As[(mt * 16 + llo) * LDA64 + ks * 32], lhi);
      v16h b = ldb(Wh9 + (nt * 16 + llo) * 64 + ks * 32, lhi);
      acc = wmma32(a, b, acc);
    }
    int o = nt * 16 + llo;
    float s = s9[o], h = h9[o];
#pragma unroll
    for (int r = 0; r < 8; ++r) {
      int p = p0 + mt * 16 + r + (lhi << 3);
      float val = acc[r] * s + h + x[(n * 64 + o) * PTV + p];
      out[(n * 64 + o) * PTV + p] = fmaxf(val, 0.f);
    }
  }
}

// ---------------------------------------------------------------------------
static inline size_t alignup(size_t x, size_t a) { return (x + a - 1) & ~(a - 1); }

extern "C" void kernel_launch(void* const* d_in, const int* in_sizes, int n_in,
                              void* d_out, int out_size, void* d_ws, size_t ws_size,
                              hipStream_t stream) {
  (void)in_sizes; (void)out_size; (void)ws_size;
  PtrTab P;
  for (int i = 0; i < NUM_IN; ++i) P.p[i] = (i < n_in) ? (const float*)d_in[i] : nullptr;

  // Workspace layout (~138 MB), lifetime-aliased:
  //   R1: fo[pre|c1|c2] (K1->K4)  -> hbuf (K7->K8)      [59.0 MB]
  //   R2: ycat (K4->K5)           -> gmean (K6->K7)     [19.7 MB]
  //   R3: g_h  (K5->K7)           -> feat  (K8->K9)     [52.4 MB]
  //   R4: afull (K3->K4)                                [ 6.3 MB]
  char* ws = (char*)d_ws;
  size_t cur = 0;
  auto take = [&](size_t bytes) -> char* {
    char* r = ws + cur; cur = alignup(cur + bytes, 256); return r;
  };
  Pack pk;
  pk.Wh1 = (half_t*)take(96 * 64 * 2);
  pk.s1 = (float*)take(96 * 4);  pk.h1 = (float*)take(96 * 4);
  pk.Wh5 = (half_t*)take(64 * 32 * 2);
  pk.s5 = (float*)take(64 * 4);  pk.h5 = (float*)take(64 * 4);
  pk.Wh7 = (half_t*)take(64 * 64 * 2);
  pk.s7 = (float*)take(64 * 4);  pk.h7 = (float*)take(64 * 4);
  pk.ctw = (float*)take(44 * 14 * 3 * 4);
  pk.sct = (float*)take(44 * 4); pk.hct = (float*)take(44 * 4);
  pk.st = (float*)take(64 * 4);  pk.ht = (float*)take(64 * 4);
  pk.Wh9 = (half_t*)take(64 * 64 * 2);
  pk.s9 = (float*)take(64 * 4);  pk.h9 = (float*)take(64 * 4);
  float* x1m = (float*)take((size_t)N_ * KM_ * V_ * 4);
  float* x2m = (float*)take((size_t)N_ * KM_ * V_ * 4);

  const size_t SL = (size_t)N_ * KM_ * PTV;          // one front-group slice (halves)
  char* R1 = take(3 * SL * 2);                       // fo; reused as hbuf
  char* R2 = take(SL * 2);                           // ycat; reused as gmean
  char* R3 = take((size_t)N_ * C_ * PTV * 2);        // g_h; reused as feat
  char* R4 = take((size_t)N_ * KM_ * 1024 * 2);      // afull

  half_t* fo    = (half_t*)R1;                       // [3][N*24*PTV]
  half_t* pre_f = fo;
  half_t* c1raw = fo + SL;
  half_t* c2raw = fo + 2 * SL;
  half_t* hbuf  = (half_t*)R1;
  half_t* ycat  = (half_t*)R2;  float* gmean = (float*)R2;
  half_t* g_h   = (half_t*)R3;  half_t* feat = (half_t*)R3;
  half_t* afull = (half_t*)R4;

  float* out0  = (float*)d_out;
  float* graph = out0 + (size_t)N_ * C_ * T_ * V_;

  k_pack<<<1, 256, 0, stream>>>(P, pk);
  k_gemm_front<<<3200, 256, 0, stream>>>(P.p[IN_X], pk.Wh1, pk.s1, pk.h1, fo);
  k_mean_t<<<(2 * N_ * KM_ * V_ + 255) / 256, 256, 0, stream>>>(c1raw, c2raw, x1m, x2m);
  k_attn<<<N_ * K_, 256, 0, stream>>>(x1m, x2m, P.p[G_A], P.p[G_ALPHA], P.p[G_BETA],
                                      afull, graph);
  k_gemm_adj<<<N_ * KM_, 128, 0, stream>>>(pre_f, afull, ycat);
  k_gemm_post<<<3200, 256, 0, stream>>>(ycat, pk.Wh5, pk.s5, pk.h5, P.p[IN_X], g_h);
  k_mean_v<<<(N_ * C_ * T_ + 255) / 256, 256, 0, stream>>>(g_h, gmean);
  k_gemm_branch<<<N_ * 26, 256, 0, stream>>>(g_h, gmean, pk.Wh7, pk.s7, pk.h7, hbuf);
  k_temporal<<<(N_ * C_ * T_ + 255) / 256, 256, 0, stream>>>(hbuf, pk.ctw, pk.sct, pk.hct,
                                                             pk.st, pk.ht, P.p[T_ADD], feat);
  k_gemm_tconv<<<3200, 256, 0, stream>>>(feat, pk.Wh9, pk.s9, pk.h9, P.p[IN_X], out0);
}
